// MultiScaleSparseTransformerBlock_29643864277320
// MI455X (gfx1250) — matrix operand order repaired and checked by reading, not compiled, not run on hardware
//
#include <hip/hip_runtime.h>
#include <hip/hip_bf16.h>

// ---------------------------------------------------------------------------
// MultiScaleSparseTransformerBlock for MI455X (gfx1250, wave32, WMMA)
//   C=256, H=W=64, HW=4096, S=4608 tokens, 8 heads x dh=32, band +-230
// GEMM-shaped math -> v_wmma_f32_16x16x32_bf16; global->LDS staging uses
// GLOBAL_LOAD_ASYNC_TO_LDS_B128 with double-buffered LDS tiles (ASYNCcnt).
// ---------------------------------------------------------------------------

#define C_DIM   256
#define HW_TOK  4096
#define S_TOK   4608
#define NHEAD   8
#define DHEAD   32
#define BAND    230

typedef __attribute__((ext_vector_type(16))) __bf16 v16bf;
typedef __attribute__((ext_vector_type(2)))  __bf16 v2bf;
typedef __attribute__((ext_vector_type(8)))  float  v8f;

#define ASYNC_LDS_B128(lds_off, gaddr)                                   \
  asm volatile("global_load_async_to_lds_b128 %0, %1, off" ::"v"(lds_off), \
               "v"(gaddr)                                                 \
               : "memory")
#define WAIT_ASYNC() asm volatile("s_wait_asynccnt 0" ::: "memory")

// ---------------- WMMA helpers (ISA 7.12.2 layouts, wave32) ----------------

static __device__ __forceinline__ v8f wmma_bf16(v16bf a, v16bf b, v8f c) {
  // (neg_a, A, neg_b, B, c_mod, C, reuse_a, reuse_b)
  return __builtin_amdgcn_wmma_f32_16x16x32_bf16(false, a, false, b,
                                                 (short)0, c, false, false);
}

// A fragment: 16x32 bf16 tile, row-major [m][k].
// lane L: m = L&15, half = L>>4; VGPR v<4: K = half*8+2v; v>=4: K = 16+half*8+2(v-4)
static __device__ __forceinline__ v16bf ld_fragA(const __bf16* t, int ld) {
  int lane = (int)(threadIdx.x & 31);
  int row = lane & 15, hl = lane >> 4;
  v16bf a;
#pragma unroll
  for (int v = 0; v < 8; ++v) {
    int kb = (v < 4) ? (hl * 8 + 2 * v) : (16 + hl * 8 + 2 * (v - 4));
    v2bf p = *(const v2bf*)(t + row * ld + kb);
    a[2 * v] = p[0];
    a[2 * v + 1] = p[1];
  }
  return a;
}

// B fragment from an [n][k] tile (row-major N x K).
// lane L: n = L&15, half = L>>4; VGPR v<4: K = half*8+2v; v>=4: K = 16+half*8+2(v-4)
static __device__ __forceinline__ v16bf ld_fragB_nk(const __bf16* t, int ld) {
  int lane = (int)(threadIdx.x & 31);
  int n = lane & 15, hl = lane >> 4;
  v16bf b;
#pragma unroll
  for (int v = 0; v < 8; ++v) {
    int k = (v < 4) ? (hl * 8 + 2 * v) : (16 + hl * 8 + 2 * (v - 4));
    v2bf p = *(const v2bf*)(t + n * ld + k);
    b[2 * v] = p[0];
    b[2 * v + 1] = p[1];
  }
  return b;
}

static __device__ __forceinline__ float wred_sum(float v) {
#pragma unroll
  for (int m = 16; m > 0; m >>= 1) v += __shfl_xor(v, m, 32);
  return v;
}

// ---------------- generic bf16 GEMM: C[M,N] = A[M,K] * B[N,K]^T + bias -----

#define BM 128
#define BN 64
#define BK 32
#define LDT 48   // LDS row stride (elems); 96B rows keep b128 LDS ops 16B-aligned

#define EPI_GELU 1

__global__ __launch_bounds__(256) void k_gemm(
    const __bf16* __restrict__ A, const __bf16* __restrict__ B,
    const float* __restrict__ bias, float* __restrict__ Cf,
    __bf16* __restrict__ Cb, int M, int N, int K, int flags) {
  __shared__ __attribute__((aligned(16))) __bf16 sA[2][BM * LDT];
  __shared__ __attribute__((aligned(16))) __bf16 sB[2][BN * LDT];
  const int bm = blockIdx.x * BM;
  const int bn = blockIdx.y * BN;
  const int t = (int)threadIdx.x;
  const int w = t >> 5, lane = t & 31;

  // staging coords: each thread owns one 16B chunk per tile section
  const int ra = t >> 2;            // 0..63
  const int ca = (t & 3) * 8;       // 0,8,16,24
  const __bf16* gA0 = A + (size_t)(bm + ra) * K + ca;
  const __bf16* gA1 = A + (size_t)(bm + 64 + ra) * K + ca;
  const __bf16* gB0 = B + (size_t)(bn + ra) * K + ca;

  auto stage = [&](int buf, int k0) {
    unsigned la0 = (unsigned)(size_t)&sA[buf][ra * LDT + ca];
    unsigned la1 = (unsigned)(size_t)&sA[buf][(64 + ra) * LDT + ca];
    unsigned lb0 = (unsigned)(size_t)&sB[buf][ra * LDT + ca];
    unsigned long long g0 = (unsigned long long)(size_t)(gA0 + k0);
    unsigned long long g1 = (unsigned long long)(size_t)(gA1 + k0);
    unsigned long long g2 = (unsigned long long)(size_t)(gB0 + k0);
    ASYNC_LDS_B128(la0, g0);
    ASYNC_LDS_B128(la1, g1);
    ASYNC_LDS_B128(lb0, g2);
  };

  v8f acc0 = {}, acc1 = {}, acc2 = {}, acc3 = {};

  stage(0, 0);
  if (BK < K) {  // warm the caches for the next tile
    __builtin_prefetch(gA0 + BK, 0, 3);
    __builtin_prefetch(gB0 + BK, 0, 3);
  }
  WAIT_ASYNC();
  __syncthreads();

  const int nk = K / BK;
  for (int ki = 0; ki < nk; ++ki) {
    const int cur = ki & 1;
    if (ki + 1 < nk) stage(cur ^ 1, (ki + 1) * BK);

    v16bf af = ld_fragA(&sA[cur][(w * 16) * LDT], LDT);
    v16bf b0 = ld_fragB_nk(&sB[cur][0 * 16 * LDT], LDT);
    v16bf b1 = ld_fragB_nk(&sB[cur][1 * 16 * LDT], LDT);
    v16bf b2 = ld_fragB_nk(&sB[cur][2 * 16 * LDT], LDT);
    v16bf b3 = ld_fragB_nk(&sB[cur][3 * 16 * LDT], LDT);
    acc0 = wmma_bf16(af, b0, acc0);
    acc1 = wmma_bf16(af, b1, acc1);
    acc2 = wmma_bf16(af, b2, acc2);
    acc3 = wmma_bf16(af, b3, acc3);

    if (ki + 1 < nk) {
      WAIT_ASYNC();       // my async writes to the next buffer are done
      __syncthreads();    // everyone done writing next / reading current
    }
  }

  // epilogue: wave w owns rows [bm+w*16, +16), cols [bn, bn+64)
  const int hl = lane >> 4, n0 = lane & 15;
  v8f* accs[4] = {&acc0, &acc1, &acc2, &acc3};
#pragma unroll
  for (int s = 0; s < 4; ++s) {
    v8f a = *accs[s];
    int n = bn + s * 16 + n0;
    float bs = bias[n];
#pragma unroll
    for (int r = 0; r < 8; ++r) {
      int m = bm + w * 16 + hl * 8 + r;
      float v = a[r] + bs;
      if (flags & EPI_GELU) v = 0.5f * v * (1.0f + erff(v * 0.70710678118654752f));
      if (Cf) Cf[(size_t)m * N + n] = v;
      if (Cb) Cb[(size_t)m * N + n] = (__bf16)v;
    }
  }
}

// ---------------- elementwise / LN / packing kernels -----------------------

__global__ void k_cvt_bf16(const float* __restrict__ in, __bf16* __restrict__ out,
                           int n) {
  for (int i = blockIdx.x * blockDim.x + threadIdx.x; i < n;
       i += gridDim.x * blockDim.x)
    out[i] = (__bf16)in[i];
}

// local tokens: LN(x_flat + local_pos); one wave per token (wave32)
__global__ __launch_bounds__(256) void k_local_ln(
    const float* __restrict__ x, const float* __restrict__ pos,
    const float* __restrict__ g, const float* __restrict__ b,
    __bf16* __restrict__ feats) {
  int w = threadIdx.x >> 5, lane = threadIdx.x & 31;
  int tok = blockIdx.x * 8 + w;
  if (tok >= HW_TOK) return;
  float v[8];
  float s = 0.f;
#pragma unroll
  for (int j = 0; j < 8; ++j) {
    int c = lane + j * 32;
    v[j] = x[(size_t)c * HW_TOK + tok] + pos[c];
    s += v[j];
  }
  float mean = wred_sum(s) * (1.0f / C_DIM);
  float q = 0.f;
#pragma unroll
  for (int j = 0; j < 8; ++j) {
    float d = v[j] - mean;
    q += d * d;
  }
  float inv = rsqrtf(wred_sum(q) * (1.0f / C_DIM) + 1e-5f);
#pragma unroll
  for (int j = 0; j < 8; ++j) {
    int c = lane + j * 32;
    feats[(size_t)tok * C_DIM + c] = (__bf16)((v[j] - mean) * inv * g[c] + b[c]);
  }
}

// row-major LN over C=256; optional additive pos; one wave per row
__global__ __launch_bounds__(256) void k_row_ln(
    const float* __restrict__ in, const float* __restrict__ pos,
    const float* __restrict__ g, const float* __restrict__ b,
    __bf16* __restrict__ out, int rows) {
  int w = threadIdx.x >> 5, lane = threadIdx.x & 31;
  int row = blockIdx.x * 8 + w;
  if (row >= rows) return;
  float v[8];
  float s = 0.f;
#pragma unroll
  for (int j = 0; j < 8; ++j) {
    int c = lane + j * 32;
    float p = pos ? pos[c] : 0.0f;
    v[j] = in[(size_t)row * C_DIM + c] + p;
    s += v[j];
  }
  float mean = wred_sum(s) * (1.0f / C_DIM);
  float q = 0.f;
#pragma unroll
  for (int j = 0; j < 8; ++j) {
    float d = v[j] - mean;
    q += d * d;
  }
  float inv = rsqrtf(wred_sum(q) * (1.0f / C_DIM) + 1e-5f);
#pragma unroll
  for (int j = 0; j < 8; ++j) {
    int c = lane + j * 32;
    out[(size_t)row * C_DIM + c] = (__bf16)((v[j] - mean) * inv * g[c] + b[c]);
  }
}

// ps=4 patches -> xp4[256][4096] bf16 (pool matrix is identity at N=256)
__global__ void k_patch4(const float* __restrict__ x, __bf16* __restrict__ xp) {
  for (int i = blockIdx.x * blockDim.x + threadIdx.x; i < 256 * 4096;
       i += gridDim.x * blockDim.x) {
    int n = i >> 12, d = i & 4095;
    int c = d >> 4, py = (d >> 2) & 3, px = d & 3;
    int bh = n >> 4, bw = n & 15;
    xp[i] = (__bf16)x[(size_t)c * HW_TOK + (bh * 4 + py) * 64 + bw * 4 + px];
  }
}

// ps=8 patches, pooled (each of 64 patch rows replicated 4x) -> xp8[256][16384]
__global__ void k_patch8(const float* __restrict__ x, __bf16* __restrict__ xp) {
  for (int i = blockIdx.x * blockDim.x + threadIdx.x; i < 256 * 16384;
       i += gridDim.x * blockDim.x) {
    int r = i >> 14, d = i & 16383;
    int c = d >> 6, py = (d >> 3) & 7, px = d & 7;
    int n = r >> 2;
    int bh = n >> 3, bw = n & 7;
    xp[i] = (__bf16)x[(size_t)c * HW_TOK + (bh * 8 + py) * 64 + bw * 8 + px];
  }
}

// qkv[S][768] f32 -> Q,K bf16 [head][S][32]; V TRANSPOSED bf16 [head][32][S].
// Q pre-scaled by 1/sqrt(dh).
__global__ void k_split(const float* __restrict__ qkv, __bf16* __restrict__ Q,
                        __bf16* __restrict__ K, __bf16* __restrict__ V) {
  const float qs = 0.17677669529663687f;
  for (int i = blockIdx.x * blockDim.x + threadIdx.x; i < S_TOK * 768;
       i += gridDim.x * blockDim.x) {
    int s = i / 768, c = i - s * 768;
    int which = c >> 8, cc = c & 255;
    int h = cc >> 5, d = cc & 31;
    float v = qkv[i];
    if (which == 0)
      Q[((size_t)h * S_TOK + s) * DHEAD + d] = (__bf16)(v * qs);
    else if (which == 1)
      K[((size_t)h * S_TOK + s) * DHEAD + d] = (__bf16)v;
    else
      V[((size_t)h * DHEAD + d) * S_TOK + s] = (__bf16)v;
  }
}

// out[c][hw] = x + attn_out[hw][c] + mlp2[hw][c]
__global__ void k_final(const float* __restrict__ x,
                        const float* __restrict__ attn_out,
                        const float* __restrict__ mlp2,
                        float* __restrict__ out) {
  for (int i = blockIdx.x * blockDim.x + threadIdx.x; i < C_DIM * HW_TOK;
       i += gridDim.x * blockDim.x) {
    int c = i >> 12, t = i & 4095;
    out[i] = x[i] + attn_out[(size_t)t * C_DIM + c] + mlp2[(size_t)t * C_DIM + c];
  }
}

// ---------------- banded flash attention (WMMA QK^T and P.V) ---------------
// block: 128 threads (4 waves); blockIdx.x = 64-row q tile, blockIdx.y = head
// K chunk staged as [k][dh]; V chunk staged as [dh][k] (V is pre-transposed),
// both via async-to-LDS with double buffering.

#define KLD 40  // LDS stride for K/V tiles: 80B rows, b128 ops stay 16B aligned

__global__ __launch_bounds__(128) void k_attn(
    const __bf16* __restrict__ Qh, const __bf16* __restrict__ Kh,
    const __bf16* __restrict__ Vt, __bf16* __restrict__ ctx) {
  __shared__ __attribute__((aligned(16))) __bf16 sK[2][32 * KLD];
  __shared__ __attribute__((aligned(16))) __bf16 sV[2][32 * KLD];
  __shared__ __attribute__((aligned(16))) __bf16 sP[4 * 16 * 32];

  const int t = (int)threadIdx.x;
  const int w = t >> 5, lane = t & 31;
  const int hl = lane >> 4, n0 = lane & 15;
  const int qt = blockIdx.x * 64;
  const int h = blockIdx.y;
  const int qr = qt + w * 16;

  const __bf16* Qb = Qh + (size_t)h * S_TOK * DHEAD;
  const __bf16* Kb = Kh + (size_t)h * S_TOK * DHEAD;
  const __bf16* Vb = Vt + (size_t)h * DHEAD * S_TOK;  // [dh][S]

  // staging coords: 128 threads x 16B cover one 32x32 bf16 tile
  const int rr = t >> 2;        // 0..31
  const int cc = (t & 3) * 8;   // 0,8,16,24

  auto stage = [&](int buf, int kc) {
    unsigned lk = (unsigned)(size_t)&sK[buf][rr * KLD + cc];
    unsigned lv = (unsigned)(size_t)&sV[buf][rr * KLD + cc];
    unsigned long long gk =
        (unsigned long long)(size_t)(Kb + (size_t)(kc + rr) * DHEAD + cc);
    unsigned long long gv =
        (unsigned long long)(size_t)(Vb + (size_t)rr * S_TOK + kc + cc);
    ASYNC_LDS_B128(lk, gk);
    ASYNC_LDS_B128(lv, gv);
  };

  // Q fragment (A-layout), invariant over the k loop
  v16bf qf;
  {
    int row = lane & 15;
#pragma unroll
    for (int v = 0; v < 8; ++v) {
      int kb = (v < 4) ? (hl * 8 + 2 * v) : (16 + hl * 8 + 2 * (v - 4));
      v2bf p = *(const v2bf*)(Qb + (size_t)(qr + row) * DHEAD + kb);
      qf[2 * v] = p[0];
      qf[2 * v + 1] = p[1];
    }
  }

  v8f o0 = {}, o1 = {};
  float mrun[8], lrun[8];
#pragma unroll
  for (int r = 0; r < 8; ++r) {
    mrun[r] = -1e30f;
    lrun[r] = 0.f;
  }

  int kmin = qt - BAND;
  if (kmin < 0) kmin = 0;
  kmin &= ~31;
  int kmax = qt + 63 + BAND + 1;
  if (kmax > S_TOK) kmax = S_TOK;
  kmax = (kmax + 31) & ~31;
  const int nch = (kmax - kmin) >> 5;

  stage(0, kmin);
  WAIT_ASYNC();
  __syncthreads();

  for (int ci = 0; ci < nch; ++ci) {
    const int kc = kmin + ci * 32;
    const int cur = ci & 1;
    if (ci + 1 < nch) stage(cur ^ 1, kc + 32);

    // scores: S0 covers k-local 0..15, S1 covers 16..31   (K-dim = dh = 32)
    v8f z = {};
    v8f s0 = wmma_bf16(qf, ld_fragB_nk(&sK[cur][0], KLD), z);
    v8f s1 = wmma_bf16(qf, ld_fragB_nk(&sK[cur][16 * KLD], KLD), z);

    // band mask + online softmax; row m = hl*8 + r, col n = n0 (D layout)
    __bf16* pw = &sP[w * 512];
#pragma unroll
    for (int r = 0; r < 8; ++r) {
      int q = qr + hl * 8 + r;
      int d0 = q - (kc + n0);      if (d0 < 0) d0 = -d0;
      int d1 = q - (kc + 16 + n0); if (d1 < 0) d1 = -d1;
      float sa = (d0 > BAND) ? -1e30f : s0[r];
      float sb = (d1 > BAND) ? -1e30f : s1[r];
      float mx = fmaxf(sa, sb);
#pragma unroll
      for (int msk = 8; msk > 0; msk >>= 1) mx = fmaxf(mx, __shfl_xor(mx, msk, 32));
      float mnew = fmaxf(mrun[r], mx);
      float pa = (mnew < -1e29f) ? 0.f : __expf(sa - mnew);
      float pb = (mnew < -1e29f) ? 0.f : __expf(sb - mnew);
      float scl = __expf(mrun[r] - mnew);
      float rs = pa + pb;
#pragma unroll
      for (int msk = 8; msk > 0; msk >>= 1) rs += __shfl_xor(rs, msk, 32);
      lrun[r] = lrun[r] * scl + rs;
      mrun[r] = mnew;
      o0[r] *= scl;
      o1[r] *= scl;
      pw[(hl * 8 + r) * 32 + n0] = (__bf16)pa;
      pw[(hl * 8 + r) * 32 + 16 + n0] = (__bf16)pb;
    }
    // wave-local: staging writes -> A-fragment reads (same wave's LDS region)
    asm volatile("s_wait_dscnt 0" ::: "memory");
    v16bf pf = ld_fragA(pw, 32);
    // P.V : B fragment rows are dh columns (V staged [dh][k])
    o0 = wmma_bf16(pf, ld_fragB_nk(&sV[cur][0], KLD), o0);
    o1 = wmma_bf16(pf, ld_fragB_nk(&sV[cur][16 * KLD], KLD), o1);

    if (ci + 1 < nch) WAIT_ASYNC();
    __syncthreads();
  }

  // normalize and scatter into ctx[S][C] at columns h*32 + d
#pragma unroll
  for (int r = 0; r < 8; ++r) {
    int m = qr + hl * 8 + r;
    float inv = (lrun[r] > 0.f) ? (1.0f / lrun[r]) : 0.f;
    ctx[(size_t)m * C_DIM + h * DHEAD + n0] = (__bf16)(o0[r] * inv);
    ctx[(size_t)m * C_DIM + h * DHEAD + 16 + n0] = (__bf16)(o1[r] * inv);
  }
}

// ---------------------------------------------------------------------------

extern "C" void kernel_launch(void* const* d_in, const int* in_sizes, int n_in,
                              void* d_out, int out_size, void* d_ws,
                              size_t ws_size, hipStream_t stream) {
  (void)in_sizes; (void)n_in; (void)out_size; (void)ws_size;

  const float* x        = (const float*)d_in[0];
  const float* lpos     = (const float*)d_in[1];
  const float* rpos0    = (const float*)d_in[2];
  const float* rpos1    = (const float*)d_in[3];
  const float* lnl_g    = (const float*)d_in[4];
  const float* lnl_b    = (const float*)d_in[5];
  const float* lnr0_g   = (const float*)d_in[6];
  const float* lnr0_b   = (const float*)d_in[7];
  const float* lnr1_g   = (const float*)d_in[8];
  const float* lnr1_b   = (const float*)d_in[9];
  const float* adj0_w   = (const float*)d_in[10];
  const float* adj0_b   = (const float*)d_in[11];
  const float* adj1_w   = (const float*)d_in[12];
  const float* adj1_b   = (const float*)d_in[13];
  const float* inproj_w = (const float*)d_in[14];
  const float* inproj_b = (const float*)d_in[15];
  const float* out_w    = (const float*)d_in[16];
  const float* out_b    = (const float*)d_in[17];
  const float* lno_g    = (const float*)d_in[18];
  const float* lno_b    = (const float*)d_in[19];
  const float* mlp_w1   = (const float*)d_in[20];
  const float* mlp_b1   = (const float*)d_in[21];
  const float* mlp_w2   = (const float*)d_in[22];
  const float* mlp_b2   = (const float*)d_in[23];
  float* out = (float*)d_out;

  char* wsb = (char*)d_ws;
  size_t off = 0;
  auto take = [&](size_t bytes) -> char* {
    size_t a = (off + 255) & ~(size_t)255;
    off = a + bytes;
    return wsb + a;
  };

  __bf16* bf_adj0 = (__bf16*)take((size_t)256 * 4096 * 2);
  __bf16* bf_adj1 = (__bf16*)take((size_t)256 * 16384 * 2);
  __bf16* bf_inpj = (__bf16*)take((size_t)768 * 256 * 2);
  __bf16* bf_outw = (__bf16*)take((size_t)256 * 256 * 2);
  __bf16* bf_w1   = (__bf16*)take((size_t)1024 * 256 * 2);
  __bf16* bf_w2   = (__bf16*)take((size_t)256 * 1024 * 2);
  __bf16* xp4     = (__bf16*)take((size_t)256 * 4096 * 2);
  __bf16* xp8     = (__bf16*)take((size_t)256 * 16384 * 2);
  __bf16* feats   = (__bf16*)take((size_t)S_TOK * C_DIM * 2);
  float*  regpre0 = (float*)take((size_t)256 * 256 * 4);
  float*  regpre1 = (float*)take((size_t)256 * 256 * 4);
  float*  qkv     = (float*)take((size_t)S_TOK * 768 * 4);
  __bf16* Q       = (__bf16*)take((size_t)NHEAD * S_TOK * DHEAD * 2);
  __bf16* Km      = (__bf16*)take((size_t)NHEAD * S_TOK * DHEAD * 2);
  __bf16* Vtr     = (__bf16*)take((size_t)NHEAD * S_TOK * DHEAD * 2);
  __bf16* ctx     = (__bf16*)take((size_t)S_TOK * C_DIM * 2);
  float*  attn_o  = (float*)take((size_t)S_TOK * C_DIM * 4);
  __bf16* ln2     = (__bf16*)take((size_t)HW_TOK * C_DIM * 2);
  __bf16* h1      = (__bf16*)take((size_t)HW_TOK * 1024 * 2);
  float*  mlp2o   = (float*)take((size_t)HW_TOK * C_DIM * 4);

  auto ew_grid = [](int n) { return dim3((unsigned)((n + 2047) >> 11)); };

  // 1) weight conversions to bf16
  k_cvt_bf16<<<ew_grid(256 * 4096), 256, 0, stream>>>(adj0_w, bf_adj0, 256 * 4096);
  k_cvt_bf16<<<ew_grid(256 * 16384), 256, 0, stream>>>(adj1_w, bf_adj1, 256 * 16384);
  k_cvt_bf16<<<ew_grid(768 * 256), 256, 0, stream>>>(inproj_w, bf_inpj, 768 * 256);
  k_cvt_bf16<<<ew_grid(256 * 256), 256, 0, stream>>>(out_w, bf_outw, 256 * 256);
  k_cvt_bf16<<<ew_grid(1024 * 256), 256, 0, stream>>>(mlp_w1, bf_w1, 1024 * 256);
  k_cvt_bf16<<<ew_grid(256 * 1024), 256, 0, stream>>>(mlp_w2, bf_w2, 256 * 1024);

  // 2) local tokens: LN(x_flat + local_pos) -> feats[0:4096]
  k_local_ln<<<HW_TOK / 8, 256, 0, stream>>>(x, lpos, lnl_g, lnl_b, feats);

  // 3) patch extraction (pool folded in)
  k_patch4<<<ew_grid(256 * 4096), 256, 0, stream>>>(x, xp4);
  k_patch8<<<ew_grid(256 * 16384), 256, 0, stream>>>(x, xp8);

  // 4) regional projections (WMMA GEMMs) + LN -> feats[4096:4608]
  k_gemm<<<dim3(2, 4), 256, 0, stream>>>(xp4, bf_adj0, adj0_b, regpre0, nullptr,
                                         256, 256, 4096, 0);
  k_gemm<<<dim3(2, 4), 256, 0, stream>>>(xp8, bf_adj1, adj1_b, regpre1, nullptr,
                                         256, 256, 16384, 0);
  k_row_ln<<<32, 256, 0, stream>>>(regpre0, rpos0, lnr0_g, lnr0_b,
                                   feats + (size_t)4096 * C_DIM, 256);
  k_row_ln<<<32, 256, 0, stream>>>(regpre1, rpos1, lnr1_g, lnr1_b,
                                   feats + (size_t)4352 * C_DIM, 256);

  // 5) qkv projection (WMMA) + head split (V transposed)
  k_gemm<<<dim3(S_TOK / BM, 768 / BN), 256, 0, stream>>>(
      feats, bf_inpj, inproj_b, qkv, nullptr, S_TOK, 768, 256, 0);
  k_split<<<ew_grid(S_TOK * 768), 256, 0, stream>>>(qkv, Q, Km, Vtr);

  // 6) banded flash attention (WMMA QK^T and P.V)
  k_attn<<<dim3(S_TOK / 64, NHEAD), 128, 0, stream>>>(Q, Km, Vtr, ctx);

  // 7) output projection (WMMA)
  k_gemm<<<dim3(S_TOK / BM, 256 / BN), 256, 0, stream>>>(
      ctx, bf_outw, out_b, attn_o, nullptr, S_TOK, 256, 256, 0);

  // 8) MLP over the 4096 local tokens: LN -> GEMM(+GELU) -> GEMM
  k_row_ln<<<HW_TOK / 8, 256, 0, stream>>>(attn_o, nullptr, lno_g, lno_b, ln2,
                                           HW_TOK);
  k_gemm<<<dim3(HW_TOK / BM, 1024 / BN), 256, 0, stream>>>(
      ln2, bf_w1, mlp_b1, nullptr, h1, HW_TOK, 1024, 256, EPI_GELU);
  k_gemm<<<dim3(HW_TOK / BM, 256 / BN), 256, 0, stream>>>(
      h1, bf_w2, mlp_b2, mlp2o, nullptr, HW_TOK, 256, 1024, 0);

  // 9) residual + transpose back to (C, H, W)
  k_final<<<ew_grid(C_DIM * HW_TOK), 256, 0, stream>>>(x, attn_o, mlp2o, out);
}
